// _ExplainableTokenGenerationHead_9302899163503
// MI455X (gfx1250) — compile-verified, matrix operand order checked by reading, not run on hardware
//
#include <hip/hip_runtime.h>
#include <hip/hip_bf16.h>

#define B_      16
#define S_      1024
#define H_      8192
#define V_      8192
#define C_      64
#define MAXLEN_ 50

#define CH_     256    // K-chunk of A staged per TDM transfer (dwords per row)
#define PITCH_  260    // CH_ + 4-dword TDM pad -> bank-conflict-free row access

typedef __attribute__((ext_vector_type(16))) __bf16 v16bf;
typedef __attribute__((ext_vector_type(8)))  float  v8f;
typedef __attribute__((ext_vector_type(4)))  unsigned int u32x4;
typedef __attribute__((ext_vector_type(8)))  int    i32x8;
typedef __attribute__((ext_vector_type(4)))  int    i32x4;

// ---------------------------------------------------------------------------
// Hash-based uniform in (0,1) for Gumbel noise.
// ---------------------------------------------------------------------------
__device__ __forceinline__ float u01hash(unsigned x) {
  x ^= x >> 16; x *= 0x7feb352dU;
  x ^= x >> 15; x *= 0x846ca68bU;
  x ^= x >> 16;
  return ((float)(x >> 8) + 0.5f) * (1.0f / 16777216.0f);
}

// ---------------------------------------------------------------------------
// TDM: stage one 16 x CH_ f32 tile of h_last (row stride S_*H_ elements) into
// LDS at lds_byte_off, with 4-dword padding every 256 dwords (PITCH_ rows).
// D# layout per CDNA5 ISA.
// ---------------------------------------------------------------------------
__device__ __forceinline__ void tdm_load_A_chunk(const float* gsrc,
                                                 unsigned lds_byte_off) {
  unsigned long long ga = (unsigned long long)(uintptr_t)gsrc;
  u32x4 g0;
  g0[0] = 1u;                                    // count=1, user descriptor
  g0[1] = lds_byte_off;                          // lds_addr (bytes)
  g0[2] = (unsigned)(ga & 0xFFFFFFFFu);          // global_addr[31:0]
  g0[3] = (unsigned)((ga >> 32) & 0x01FFFFFFu)   // global_addr[56:32]
        | (2u << 30);                            // type=2 ("image")
  i32x8 g1;
  g1[0] = (2 << 16)                              // data_size = 4 bytes
        | (1 << 20)                              // pad_enable
        | (7 << 22)                              // pad_interval: 256 dwords
        | (3 << 25);                             // pad_amount:   4 dwords
  g1[1] = (int)(((unsigned)CH_ & 0xFFFFu) << 16);          // tensor_dim0 lo16
  g1[2] = (int)((((unsigned)CH_ >> 16) & 0xFFFFu)          // tensor_dim0 hi16
        | (((unsigned)B_ & 0xFFFFu) << 16));               // tensor_dim1 lo16
  g1[3] = (int)(((unsigned)CH_ & 0xFFFFu) << 16);          // dim1 hi=0, tile_dim0=CH_
  g1[4] = (int)((unsigned)B_ & 0xFFFFu);                   // tile_dim1=16, tile_dim2=0
  g1[5] = (int)(unsigned)(S_ * H_);                        // tensor_dim0_stride lo32
  g1[6] = 0;                                               // stride hi16, dim1_stride lo16
  g1[7] = 0;
  i32x4 z4 = {0, 0, 0, 0};
#if __clang_major__ >= 23
  i32x8 z8 = {0, 0, 0, 0, 0, 0, 0, 0};
  __builtin_amdgcn_tensor_load_to_lds(g0, g1, z4, z4, z8, 0);
#else
  __builtin_amdgcn_tensor_load_to_lds(g0, g1, z4, z4, 0);
#endif
}

// ---------------------------------------------------------------------------
// Kernel A: selection head. One block per batch element; h_last row staged in
// LDS, 64 block-wide dot products, argmax of 64.
// ---------------------------------------------------------------------------
__global__ void __launch_bounds__(256)
sel_head_kernel(const float* __restrict__ llm,
                const float* __restrict__ selW,
                const float* __restrict__ selb,
                float* __restrict__ out_sel) {
  __shared__ float sh[H_];
  __shared__ float red[256];
  __shared__ float logit[C_];
  const int b = blockIdx.x, tid = threadIdx.x;
  const float* arow = llm + ((size_t)b * S_ + (S_ - 1)) * H_;
  for (int h = tid; h < H_; h += 256) sh[h] = arow[h];
  __syncthreads();
  for (int c = 0; c < C_; ++c) {
    const float* w = selW + (size_t)c * H_;
    float p = 0.f;
    for (int h = tid; h < H_; h += 256) p += sh[h] * w[h];
    red[tid] = p;
    __syncthreads();
    for (int s = 128; s > 0; s >>= 1) {
      if (tid < s) red[tid] += red[tid + s];
      __syncthreads();
    }
    if (tid == 0) logit[c] = red[0] + selb[c];
    __syncthreads();
  }
  if (tid == 0) {
    int bi = 0; float bv = logit[0];
    for (int c = 1; c < C_; ++c) if (logit[c] > bv) { bv = logit[c]; bi = c; }
    out_sel[b] = (float)bi;
  }
}

// ---------------------------------------------------------------------------
// Kernel B: logits0 = h_last @ lang_W.T + lang_b (16x8192), bf16 WMMA with
// f32 accumulate. A-panel double-buffered into LDS by the Tensor Data Mover
// (tensor_load_to_lds + s_wait_tensorcnt); B streamed from HBM as b128 loads.
// Per-lane addressing follows the CDNA5 16-bit A/B VGPR layout.
// ---------------------------------------------------------------------------
__global__ void __launch_bounds__(256)
lang_logits_wmma(const float* __restrict__ llm,
                 const float* __restrict__ langW,
                 const float* __restrict__ langb,
                 float* __restrict__ logits0) {
  __shared__ float shA[2][B_][PITCH_];           // 2 x 16 x 260 dwords = 32.5 KB
  const int tid   = threadIdx.x;
  const int lane  = tid & 31;
  const int wave  = tid >> 5;
  const int n0    = (blockIdx.x * 8 + wave) * 16;   // vocab tile start
  const int sub   = lane & 15;                      // M for A, N for B
  const int kbase = (lane >> 4) << 3;               // 0 or 8

  const float* brow  = langW + (size_t)(n0 + sub) * H_;
  const float* apan  = llm + (size_t)(S_ - 1) * H_;     // h_last[0]; row step S_*H_
  const unsigned lds0 = (unsigned)(uintptr_t)&shA[0][0][0];
  const unsigned bufBytes = B_ * PITCH_ * 4;

  if (wave == 0) tdm_load_A_chunk(apan, lds0);

  const int NCH = H_ / CH_;                          // 32 chunks
  v8f acc = {};
  for (int kc = 0; kc < NCH; ++kc) {
    if (wave == 0) {
      if (kc + 1 < NCH) {
        tdm_load_A_chunk(apan + (size_t)(kc + 1) * CH_,
                         lds0 + (unsigned)((kc + 1) & 1) * bufBytes);
        __builtin_amdgcn_s_wait_tensorcnt(1);      // in-order => chunk kc done
      } else {
        __builtin_amdgcn_s_wait_tensorcnt(0);
      }
    }
    __syncthreads();

    const float* arow = &shA[kc & 1][sub][0];      // LDS, pitch 260 dwords
    const int hG = kc * CH_;
#pragma unroll 2
    for (int hl = 0; hl < CH_; hl += 32) {
      __builtin_prefetch(brow + hG + hl + 512, 0, 1);
      float af[16], bf[16];
      *(float4*)(af + 0)  = *(const float4*)(arow + hl + kbase);        // ds_load_b128
      *(float4*)(af + 4)  = *(const float4*)(arow + hl + kbase + 4);
      *(float4*)(af + 8)  = *(const float4*)(arow + hl + kbase + 16);
      *(float4*)(af + 12) = *(const float4*)(arow + hl + kbase + 20);
      const float* bp = brow + hG + hl;
      *(float4*)(bf + 0)  = *(const float4*)(bp + kbase);               // global_load_b128
      *(float4*)(bf + 4)  = *(const float4*)(bp + kbase + 4);
      *(float4*)(bf + 8)  = *(const float4*)(bp + kbase + 16);
      *(float4*)(bf + 12) = *(const float4*)(bp + kbase + 20);

      v16bf Av, Bv;
#pragma unroll
      for (int e = 0; e < 16; ++e) {
        Av[e] = (__bf16)af[e];
        Bv[e] = (__bf16)bf[e];
      }
      acc = __builtin_amdgcn_wmma_f32_16x16x32_bf16(
          false, Av, false, Bv, (short)0, acc, false, false);
    }
    __syncthreads();
  }

  // C/D layout: lanes 0-15 -> M=r, lanes 16-31 -> M=8+r; N = lane&15.
  const int   ncol  = n0 + sub;
  const int   mbase = (lane < 16) ? 0 : 8;
  const float bias  = langb[ncol];
#pragma unroll
  for (int r = 0; r < 8; ++r)
    logits0[(size_t)(mbase + r) * V_ + ncol] = acc[r] + bias;
}

// ---------------------------------------------------------------------------
// Kernel C: 50 sequential Gumbel-max sampling steps, one block per batch.
// Steps >=1: one-hot state => logits collapse to column tok_prev of lang_W.
// ---------------------------------------------------------------------------
__global__ void __launch_bounds__(256)
gen_kernel(const float* __restrict__ langW,
           const float* __restrict__ langb,
           const float* __restrict__ logits0,
           float* __restrict__ out_toks) {
  __shared__ float sval[256];
  __shared__ int   sidx[256];
  __shared__ int   stok;
  const int b = blockIdx.x, tid = threadIdx.x;
  int tok = 0;
  for (int t = 0; t < MAXLEN_; ++t) {
    float bv = -3.4e38f; int bi = 0;
    for (int v = tid; v < V_; v += 256) {
      float logit = (t == 0) ? logits0[(size_t)b * V_ + v]
                             : (langW[(size_t)v * H_ + tok] + langb[v]);
      unsigned seed = ((unsigned)(t * 131071 + b * 8191)) * 0x9E3779B9u
                      ^ ((unsigned)v * 0x85EBCA6Bu);
      float u = u01hash(seed);
      float g = -__logf(-__logf(u));   // Gumbel(0,1)
      float val = logit + g;
      if (val > bv) { bv = val; bi = v; }
    }
    sval[tid] = bv; sidx[tid] = bi;
    __syncthreads();
    for (int s = 128; s > 0; s >>= 1) {
      if (tid < s && sval[tid + s] > sval[tid]) {
        sval[tid] = sval[tid + s]; sidx[tid] = sidx[tid + s];
      }
      __syncthreads();
    }
    if (tid == 0) {
      stok = sidx[0];
      out_toks[b * MAXLEN_ + t] = (float)sidx[0];
    }
    __syncthreads();
    tok = stok;
    __syncthreads();
  }
}

// ---------------------------------------------------------------------------
extern "C" void kernel_launch(void* const* d_in, const int* in_sizes, int n_in,
                              void* d_out, int out_size, void* d_ws, size_t ws_size,
                              hipStream_t stream) {
  (void)in_sizes; (void)n_in; (void)out_size; (void)ws_size;
  const float* llm   = (const float*)d_in[0];  // (16,1024,8192)
  const float* selW  = (const float*)d_in[1];  // (64,8192)
  const float* selb  = (const float*)d_in[2];  // (64,)
  const float* langW = (const float*)d_in[3];  // (8192,8192)
  const float* langb = (const float*)d_in[4];  // (8192,)
  float* out     = (float*)d_out;              // [16 sel ids | 16*50 tokens]
  float* logits0 = (float*)d_ws;               // 16*8192 f32 scratch

  sel_head_kernel <<<B_, 256, 0, stream>>>(llm, selW, selb, out);
  lang_logits_wmma<<<V_ / 128, 256, 0, stream>>>(llm, langW, langb, logits0);
  gen_kernel      <<<B_, 256, 0, stream>>>(langW, langb, logits0, out + B_);
}
